// SelfAttentionND_55800215110249
// MI455X (gfx1250) — compile-verified, matrix operand order checked by reading, not run on hardware
//
#include <hip/hip_runtime.h>

// ---------------------------------------------------------------------------
// CDNA5 (gfx1250) WMMA self-attention (coltran-style factorized-bias causal
// attention).  All matmuls on v_wmma_f32_16x16x32_bf16; attention K/V tiles
// double-buffered in LDS via global_load_async_to_lds_b128 (ASYNCcnt) so the
// copy of tile s+1 overlaps the WMMAs of tile s.
//
// Shapes: B=4, R=C=64, S=4096, D=512, H=4, DH=128.
// ---------------------------------------------------------------------------

typedef __attribute__((ext_vector_type(16))) __bf16 v16bf;
typedef __attribute__((ext_vector_type(8)))  __bf16 v8bf;
typedef __attribute__((ext_vector_type(8)))  float  v8f;

#define BB   4
#define HH   4
#define SS   4096
#define DD   512
#define DHH  128
#define NEGMASK (-1e6f)

__device__ __forceinline__ v8f wmma_bf16(v16bf a, v16bf b, v8f c) {
  // (neg_a, A, neg_b, B, c_mod, C, reuse_a, reuse_b)
  return __builtin_amdgcn_wmma_f32_16x16x32_bf16(false, a, false, b,
                                                 (short)0, c, false, false);
}

// A/B fragment, 16-bit, 16x32: lane L holds row/col (L&15); per-lane elements
//   e=0..7  <-> K = half*8 + e
//   e=8..15 <-> K = 16 + half*8 + (e-8)        (half = L>>4)
// => fragment == two contiguous 16B chunks when operand is K-contiguous.
__device__ __forceinline__ v16bf frag_from_chunks(v8bf c0, v8bf c1) {
  v16bf f;
#pragma unroll
  for (int i = 0; i < 8; ++i) { f[i] = c0[i]; f[8 + i] = c1[i]; }
  return f;
}

__device__ __forceinline__ v16bf load_frag_rm_bf16(const __bf16* rowptr,
                                                   int kc, int half) {
  v8bf c0 = *(const v8bf*)(rowptr + kc + half * 8);
  v8bf c1 = *(const v8bf*)(rowptr + kc + 16 + half * 8);
  return frag_from_chunks(c0, c1);
}

// Async global -> LDS copy, 16 bytes per lane (ASYNCcnt-tracked).
__device__ __forceinline__ void async_copy_b128(unsigned lds_off,
                                                const __bf16* gaddr) {
  asm volatile("global_load_async_to_lds_b128 %0, %1, off"
               :: "v"(lds_off), "v"(gaddr) : "memory");
}

// Stage one 32-key step: K tile (32x128, key-major) + Vt tile (128x32,
// dh-major).  8 async b128 chunk-copies per thread (128-thread block).
__device__ __forceinline__ void stage_kv(const __bf16* ksrc, const __bf16* vsrc,
                                         unsigned kbase, unsigned vbase,
                                         int tid) {
#pragma unroll
  for (int j = 0; j < 4; ++j) {
    int c = tid + j * 128;                          // 512 x 16B chunks
    int row = c >> 3, seg = c & 7;
    async_copy_b128(kbase + (unsigned)(row * DHH + seg * 8) * 2,
                    ksrc + row * DHH + seg * 8);
  }
#pragma unroll
  for (int j = 0; j < 4; ++j) {
    int c = tid + j * 128;                          // 512 x 16B chunks
    int dh = c >> 2, seg = c & 3;
    async_copy_b128(vbase + (unsigned)(dh * 32 + seg * 8) * 2,
                    vsrc + (size_t)dh * SS + seg * 8);
  }
}

// ---------------------------------------------------------------------------
// Kernel 0: one-time transpose of the four weight matrices to bf16 [N][K].
// wq/wk/wv: [D][H*DH] -> T[H*DH][D];  wo: [H*DH][D] -> T[D][H*DH].
// ---------------------------------------------------------------------------
__global__ void transpose_weights_kernel(const float* __restrict__ wq,
                                         const float* __restrict__ wk,
                                         const float* __restrict__ wv,
                                         const float* __restrict__ wo,
                                         __bf16* __restrict__ wT) {
  const int idx = blockIdx.x * 256 + threadIdx.x;   // 4 * 512 * 512 total
  const int sel = idx >> 18;
  const int rem = idx & 262143;
  const int k   = rem >> 9;
  const int n   = rem & 511;                        // coalesced read dim
  const float* src = (sel == 0) ? wq : (sel == 1) ? wk : (sel == 2) ? wv : wo;
  wT[(size_t)sel * DD * DD + (size_t)n * DD + k] = (__bf16)src[(size_t)k * DD + n];
}

// ---------------------------------------------------------------------------
// Kernel 1: fused Q/K/V projection.  All 8 waves of a block share the same
// 16 rows of x; the A tile is staged once per k-step in LDS (f32 -> bf16
// cooperative conversion), then each wave reads its A fragment with two
// ds_load_b128.  Q,K stored [B,H,S,DH];  V stored transposed [B,H,DH,S].
// ---------------------------------------------------------------------------
__global__ void qkv_proj_kernel(const float* __restrict__ x,
                                const __bf16* __restrict__ wqT,
                                const __bf16* __restrict__ wkT,
                                const __bf16* __restrict__ wvT,
                                __bf16* __restrict__ Q,
                                __bf16* __restrict__ K,
                                __bf16* __restrict__ Vt) {
  __shared__ __bf16 Asm[16][32];                    // 1 KB shared A tile

  const int lane = threadIdx.x & 31;
  const int wave = threadIdx.x >> 5;
  const int half = lane >> 4;
  const int l16  = lane & 15;
  const int tm   = blockIdx.x >> 2;                 // [0,1024) row tile
  const int ng   = blockIdx.x & 3;                  // n-group of 128 cols
  const int n    = (ng * 8 + wave) * 16 + l16;      // [0,512)

  v8f accq = {}, acck = {}, accv = {};
  const __bf16* wqrow = wqT + (size_t)n * DD;
  const __bf16* wkrow = wkT + (size_t)n * DD;
  const __bf16* wvrow = wvT + (size_t)n * DD;
  const float*  xtile = x + (size_t)tm * 16 * DD;

#pragma unroll 1
  for (int kc = 0; kc < DD; kc += 32) {
    {                                               // cooperative A staging
      const int t = threadIdx.x;                    // 256 threads, 512 elems
      const float2 xv =
          *(const float2*)(xtile + (size_t)(t >> 4) * DD + kc + (t & 15) * 2);
      Asm[t >> 4][(t & 15) * 2]     = (__bf16)xv.x;
      Asm[t >> 4][(t & 15) * 2 + 1] = (__bf16)xv.y;
    }
    __syncthreads();
    v16bf a  = load_frag_rm_bf16(&Asm[l16][0], 0, half);
    v16bf bq = load_frag_rm_bf16(wqrow, kc, half);
    v16bf bk = load_frag_rm_bf16(wkrow, kc, half);
    v16bf bv = load_frag_rm_bf16(wvrow, kc, half);
    accq = wmma_bf16(a, bq, accq);
    acck = wmma_bf16(a, bk, acck);
    accv = wmma_bf16(a, bv, accv);
    __syncthreads();                                // Asm reusable
  }

  const float scale = 0.08838834764831845f;         // DH^-0.5
  const int h = n >> 7, e = n & (DHH - 1);
#pragma unroll
  for (int r = 0; r < 8; ++r) {
    int m = tm * 16 + r + 8 * half;                 // global row in [0, B*S)
    int b = m >> 12, s = m & (SS - 1);
    size_t o = ((size_t)(b * HH + h) * SS + s) * DHH + e;
    Q[o] = (__bf16)(accq[r] * scale);
    K[o] = (__bf16)acck[r];
    Vt[((size_t)(b * HH + h) * DHH + e) * SS + s] = (__bf16)accv[r];
  }
}

// ---------------------------------------------------------------------------
// Kernel 2: block-cooperative causal flash attention with factorized bias.
// Block = 4 waves sharing (b, h) and 64 consecutive queries.  K/Vt tiles are
// double-buffered in LDS: step s issues async copies for tile s+1, then waits
// asynccnt<=8 so only tile s's copies must have retired (async loads complete
// in order).  Step count is block-uniform; tiles past a wave's causal horizon
// are annihilated by the -1e6 mask (exp underflows to 0).
// ---------------------------------------------------------------------------
__global__ void attn_kernel(const __bf16* __restrict__ Q,
                            const __bf16* __restrict__ Kg,
                            const __bf16* __restrict__ Vt,
                            const float* __restrict__ rel_row,
                            const float* __restrict__ rel_col,
                            __bf16* __restrict__ O) {
  __shared__ __bf16 Ksm[2][32][DHH];                // 2 x 8 KB, key-major
  __shared__ __bf16 Vsm[2][DHH][32];                // 2 x 8 KB, dh-major
  __shared__ __bf16 plds[4][16 * 32];               // 4 KB, per-wave P stage

  const int lane = threadIdx.x & 31;
  const int wave = threadIdx.x >> 5;
  const int half = lane >> 4;
  const int l16  = lane & 15;
  const int qblk = blockIdx.x & 63;                 // 64-query block
  const int h    = (blockIdx.x >> 6) & (HH - 1);
  const int b    = blockIdx.x >> 8;
  const int q0   = qblk * 64 + wave * 16;

  const __bf16* Qb  = Q  + ((size_t)(b * HH + h) * SS) * DHH;
  const __bf16* Kb  = Kg + ((size_t)(b * HH + h) * SS) * DHH;
  const __bf16* Vtb = Vt + ((size_t)(b * HH + h) * DHH) * SS;

  const unsigned kbase0 = (unsigned)(size_t)&Ksm[0][0][0];
  const unsigned vbase0 = (unsigned)(size_t)&Vsm[0][0][0];

  // Q fragments: 4 x (16 rows x 32 dh), contiguous 16B loads.
  v16bf qf[4];
  const __bf16* qrow = Qb + (size_t)(q0 + l16) * DHH;
#pragma unroll
  for (int c = 0; c < 4; ++c) qf[c] = load_frag_rm_bf16(qrow, c * 32, half);

  v8f oacc[8];
#pragma unroll
  for (int t = 0; t < 8; ++t) oacc[t] = (v8f){};
  float m_[8], l_[8];
#pragma unroll
  for (int r = 0; r < 8; ++r) { m_[r] = -1e30f; l_[r] = 0.f; }

  const int nsteps = 2 * qblk + 2;                  // keys [0, qblk*64+64)

  stage_kv(Kb, Vtb, kbase0, vbase0, threadIdx.x);   // tile 0 -> buffer 0

#pragma unroll 1
  for (int step = 0; step < nsteps; ++step) {
    const int k0  = step * 32;
    const int cur = step & 1;

    // ---- issue async copies for the NEXT tile, wait for the CURRENT ----
    if (step + 1 < nsteps) {
      const unsigned nb = (unsigned)((step + 1) & 1) * 8192u;
      stage_kv(Kb + (size_t)(k0 + 32) * DHH, Vtb + (k0 + 32),
               kbase0 + nb, vbase0 + nb, threadIdx.x);
      asm volatile("s_wait_asynccnt 0x8" ::: "memory");
    } else {
      asm volatile("s_wait_asynccnt 0x0" ::: "memory");
    }
    if (step + 2 < nsteps) {                        // warm L2 two tiles out
      __builtin_prefetch((const void*)(Kb + (size_t)(k0 + 64) * DHH), 0, 1);
      __builtin_prefetch((const void*)(Vtb + k0 + 64), 0, 1);
    }
    __syncthreads();                                // current tile visible

    // ---- logits: two 16x16 subtiles over 4 dh chunks (K from LDS) -------
    v8f s0 = {}, s1 = {};
#pragma unroll
    for (int c = 0; c < 4; ++c) {
      v16bf b0 = load_frag_rm_bf16(&Ksm[cur][l16][0],      c * 32, half);
      v16bf b1 = load_frag_rm_bf16(&Ksm[cur][16 + l16][0], c * 32, half);
      s0 = wmma_bf16(qf[c], b0, s0);
      s1 = wmma_bf16(qf[c], b1, s1);
    }

    // ---- bias + causal mask + online softmax ----------------------------
    const int kp0 = k0 + l16, kp1 = k0 + 16 + l16;
    const int kr0 = kp0 >> 6, kc0 = kp0 & 63;
    const int kr1 = kp1 >> 6, kc1 = kp1 & 63;
#pragma unroll
    for (int r = 0; r < 8; ++r) {
      const int M    = r + 8 * half;
      const int qpos = q0 + M;
      const int qr   = qpos >> 6, qc = qpos & 63;
      float v0 = s0[r] + rel_row[h * 128 + 63 + kr0 - qr]
                       + rel_col[h * 128 + 63 + kc0 - qc]
                       + (kp0 > qpos ? NEGMASK : 0.f);
      float v1 = s1[r] + rel_row[h * 128 + 63 + kr1 - qr]
                       + rel_col[h * 128 + 63 + kc1 - qc]
                       + (kp1 > qpos ? NEGMASK : 0.f);

      float rmax = fmaxf(v0, v1);
#pragma unroll
      for (int off = 1; off < 16; off <<= 1)
        rmax = fmaxf(rmax, __shfl_xor(rmax, off, 32));
      const float newm = fmaxf(m_[r], rmax);
      const float p0 = __expf(v0 - newm);
      const float p1 = __expf(v1 - newm);
      const float corr = __expf(m_[r] - newm);
      m_[r] = newm;
      float rsum = p0 + p1;
#pragma unroll
      for (int off = 1; off < 16; off <<= 1)
        rsum += __shfl_xor(rsum, off, 32);
      l_[r] = l_[r] * corr + rsum;
#pragma unroll
      for (int t = 0; t < 8; ++t) oacc[t][r] *= corr;

      plds[wave][M * 32 + l16]      = (__bf16)p0;   // C-layout -> LDS
      plds[wave][M * 32 + 16 + l16] = (__bf16)p1;
    }
    asm volatile("s_wait_dscnt 0x0" ::: "memory");  // wave-private LDS RAW

    // ---- P as A-fragment from LDS, then PV against Vt tile --------------
    v16bf pf = load_frag_rm_bf16(&plds[wave][l16 * 32], 0, half);
#pragma unroll
    for (int t = 0; t < 8; ++t) {
      v16bf vf = load_frag_rm_bf16(&Vsm[cur][t * 16 + l16][0], 0, half);
      oacc[t] = wmma_bf16(pf, vf, oacc[t]);
    }
    __syncthreads();                                // tile readers done
  }

  // ---- normalize and store O as bf16 [B, S, H, DH] -----------------------
#pragma unroll
  for (int r = 0; r < 8; ++r) {
    const float inv  = 1.0f / l_[r];
    const int   qpos = q0 + r + 8 * half;
    size_t base = (((size_t)b * SS + qpos) * HH + h) * DHH;
#pragma unroll
    for (int t = 0; t < 8; ++t)
      O[base + t * 16 + l16] = (__bf16)(oacc[t][r] * inv);
  }
}

// ---------------------------------------------------------------------------
// Kernel 3: output projection  out[B*S, D] = O[B*S, H*DH] x woT[D, H*DH]^T
// ---------------------------------------------------------------------------
__global__ void out_proj_kernel(const __bf16* __restrict__ O,
                                const __bf16* __restrict__ woT,
                                float* __restrict__ out) {
  const int lane = threadIdx.x & 31;
  const int wave = threadIdx.x >> 5;
  const int half = lane >> 4;
  const int l16  = lane & 15;
  const int tm   = blockIdx.x >> 2;
  const int ng   = blockIdx.x & 3;
  const int n    = (ng * 8 + wave) * 16 + l16;

  v8f acc = {};
  const __bf16* orow = O   + (size_t)(tm * 16 + l16) * (HH * DHH);
  const __bf16* wrow = woT + (size_t)n * (HH * DHH);
#pragma unroll 1
  for (int kc = 0; kc < HH * DHH; kc += 32) {
    v16bf a  = load_frag_rm_bf16(orow, kc, half);
    v16bf bw = load_frag_rm_bf16(wrow, kc, half);
    acc = wmma_bf16(a, bw, acc);
  }
#pragma unroll
  for (int r = 0; r < 8; ++r)
    out[(size_t)(tm * 16 + r + 8 * half) * DD + n] = acc[r];
}

// ---------------------------------------------------------------------------
extern "C" void kernel_launch(void* const* d_in, const int* in_sizes, int n_in,
                              void* d_out, int out_size, void* d_ws,
                              size_t ws_size, hipStream_t stream) {
  const float* x       = (const float*)d_in[0];
  const float* wq      = (const float*)d_in[1];
  const float* wk      = (const float*)d_in[2];
  const float* wv      = (const float*)d_in[3];
  const float* wo      = (const float*)d_in[4];
  const float* rel_row = (const float*)d_in[5];
  const float* rel_col = (const float*)d_in[6];

  const size_t elems = (size_t)BB * SS * HH * DHH;  // 8M bf16 per buffer
  __bf16* Q  = (__bf16*)d_ws;
  __bf16* K  = Q  + elems;
  __bf16* Vt = K  + elems;
  __bf16* O  = Vt + elems;
  __bf16* wT = O  + elems;                          // 4 x [512 x 512] bf16
  __bf16* wqT = wT + 0 * (size_t)DD * DD;
  __bf16* wkT = wT + 1 * (size_t)DD * DD;
  __bf16* wvT = wT + 2 * (size_t)DD * DD;
  __bf16* woT = wT + 3 * (size_t)DD * DD;

  // 4 x 512 x 512 weight elements / 256 threads -> 4096 blocks
  transpose_weights_kernel<<<4096, 256, 0, stream>>>(wq, wk, wv, wo, wT);
  // 1024 row-tiles x 4 n-groups -> 4096 blocks of 8 waves
  qkv_proj_kernel<<<4096, 256, 0, stream>>>(x, wqT, wkT, wvT, Q, K, Vt);
  // B*H*(S/64) = 1024 blocks of 4 waves (64 queries per block)
  attn_kernel<<<1024, 128, 0, stream>>>(Q, K, Vt, rel_row, rel_col, O);
  // 1024 row-tiles x 4 n-groups -> 4096 blocks of 8 waves
  out_proj_kernel<<<4096, 256, 0, stream>>>(O, woT, (float*)d_out);
}
